// GPT_71631464563129
// MI455X (gfx1250) — matrix operand order repaired and checked by reading, not compiled, not run on hardware
//
#include <hip/hip_runtime.h>
#include <math.h>

#define NTOK   2048
#define DMODEL 1024
#define DFFN   4096
#define NLAYER 6
#define NHEAD  16
#define SEQ    1024
#define VOCAB  32000
#define HD     64

typedef __bf16 bf16_t;
typedef __attribute__((ext_vector_type(16))) __bf16 v16bf;
typedef __attribute__((ext_vector_type(8)))  float  v8f;
typedef __attribute__((ext_vector_type(4)))  unsigned u32x4;
typedef __attribute__((ext_vector_type(8)))  int      i32x8;
typedef __attribute__((ext_vector_type(4)))  int      i32x4;

#if __has_builtin(__builtin_amdgcn_tensor_load_to_lds)
#define HAS_TDM 1
#else
#define HAS_TDM 0
#endif

// ---------- helpers ----------
__device__ __forceinline__ bf16_t f2bf(float f) {
  unsigned u = __builtin_bit_cast(unsigned, f);
  unsigned r = u + 0x7FFFu + ((u >> 16) & 1u);   // round-to-nearest-even
  unsigned short h = (unsigned short)(r >> 16);
  return __builtin_bit_cast(bf16_t, h);
}
__device__ __forceinline__ float bf2f(bf16_t b) {
  unsigned short h = __builtin_bit_cast(unsigned short, b);
  unsigned u = ((unsigned)h) << 16;
  return __builtin_bit_cast(float, u);
}
__device__ __forceinline__ v8f wmma_bf16(v16bf a, v16bf b, v8f c) {
  return __builtin_amdgcn_wmma_f32_16x16x32_bf16(false, a, false, b,
                                                 (short)0, c, false, false);
}
// A fragment (16x32 bf16, contiguous k order, not the ISA table's): lane m=l&15 owns row m; half g=l>>4:
// VGPR i in 0..7 -> K = g*16+2i,{+1} (matches B fragment order).
__device__ __forceinline__ v16bf load_a_frag(const bf16_t* rowp, int lg) {
  union { v16bf v; unsigned u[8]; } f;
#pragma unroll
  for (int i = 0; i < 4; i++) {
    f.u[i]     = *(const unsigned*)(rowp + lg * 16 + 2 * i);
    f.u[4 + i] = *(const unsigned*)(rowp + 8 + lg * 16 + 2 * i);
  }
  return f.v;
}
// B fragment (32x16 bf16): lane n=l&15 owns col n; half g holds K=g*16..+15,
// VGPR j -> K = g*16+2j,{+1}. colp = this lane's column stored k-contiguous.
__device__ __forceinline__ v16bf load_b_frag(const bf16_t* colp, int lg) {
  union { v16bf v; unsigned u[8]; } f;
#pragma unroll
  for (int j = 0; j < 8; j++)
    f.u[j] = *(const unsigned*)(colp + lg * 16 + 2 * j);
  return f.v;
}

#if HAS_TDM
// TDM: DMA one 2D bf16 tile (64 x 64 elems, row stride in elems) into LDS.
// D# packed per CDNA5 ISA 8.3/8.4: group0 {count=1, lds_addr, global_addr, type=2},
// group1 {data_size=2B, tensor_dim0=64, tensor_dim1=rows, tile 64x64, dim0_stride}.
// This toolchain declares the 6-arg builtin: (g0, g1, g2, g3, g1b, cpol).
__device__ __forceinline__ void tdm_load_tile64x64(const bf16_t* gaddr,
                                                   unsigned lds_off,
                                                   unsigned rows,
                                                   unsigned rowstride_elems) {
  unsigned long long ga = (unsigned long long)(uintptr_t)gaddr;
  u32x4 g0;
  g0[0] = 1u;                                          // count=1 (user mode)
  g0[1] = lds_off;                                     // lds_addr (bytes)
  g0[2] = (unsigned)(ga & 0xffffffffu);                // global_addr[31:0]
  g0[3] = (unsigned)((ga >> 32) & 0x01ffffffu)         // global_addr[56:32]
        | (2u << 30);                                  // type = 2 ("image")
  i32x8 g1;
  g1[0] = (int)(1u << 16);            // wg_mask=0 | data_size=1 (2 bytes)
  g1[1] = (int)(64u << 16);           // tensor_dim0[15:0]=64  (bits 63:48)
  g1[2] = (int)(rows << 16);          // dim0 hi=0 | tensor_dim1[15:0]=rows
  g1[3] = (int)(64u << 16);           // dim1 hi=0 | tile_dim0=64
  g1[4] = (int)64u;                   // tile_dim1=64 | tile_dim2=0
  g1[5] = (int)rowstride_elems;       // tensor_dim0_stride[31:0]
  g1[6] = 0;                          // stride hi | tensor_dim1_stride lo
  g1[7] = 0;
  i32x4 z4 = {0, 0, 0, 0};
  i32x8 z8 = {0, 0, 0, 0, 0, 0, 0, 0};
  __builtin_amdgcn_tensor_load_to_lds(g0, g1, z4, z4, z8, 0);
}
#endif

// ---------- embedding ----------
__global__ __launch_bounds__(256)
void embed_kernel(const int* __restrict__ idx, const float* __restrict__ tok,
                  const float* __restrict__ pos, float* __restrict__ x) {
  int n = blockIdx.x;
  int t = n & (SEQ - 1);
  const float* te = tok + (long)idx[n] * DMODEL;
  const float* pe = pos + (long)t * DMODEL;
  float* xo = x + (long)n * DMODEL;
  for (int d = threadIdx.x; d < DMODEL; d += 256) xo[d] = te[d] + pe[d];
}

// ---------- layernorm -> bf16 ----------
__global__ __launch_bounds__(256)
void ln_bf16_kernel(const float* __restrict__ x, const float* __restrict__ w,
                    const float* __restrict__ b, bf16_t* __restrict__ out) {
  int n = blockIdx.x;
  const float* xr = x + (long)n * DMODEL;
  float s = 0.f, s2 = 0.f;
  for (int d = threadIdx.x; d < DMODEL; d += 256) {
    float v = xr[d]; s += v; s2 += v * v;
  }
#pragma unroll
  for (int off = 16; off >= 1; off >>= 1) {
    s  += __shfl_xor(s, off, 32);
    s2 += __shfl_xor(s2, off, 32);
  }
  __shared__ float red[16];
  int wave = threadIdx.x >> 5, lane = threadIdx.x & 31;
  if (lane == 0) { red[wave] = s; red[wave + 8] = s2; }
  __syncthreads();
  if (threadIdx.x == 0) {
    float a = 0.f, c = 0.f;
#pragma unroll
    for (int i = 0; i < 8; i++) { a += red[i]; c += red[i + 8]; }
    red[0] = a; red[8] = c;
  }
  __syncthreads();
  float mean = red[0] * (1.0f / DMODEL);
  float var  = red[8] * (1.0f / DMODEL) - mean * mean;
  float rstd = rsqrtf(var + 1e-5f);
  bf16_t* o = out + (long)n * DMODEL;
  for (int d = threadIdx.x; d < DMODEL; d += 256)
    o[d] = f2bf((xr[d] - mean) * rstd * w[d] + b[d]);
}

// ---------- GEMM:  C[M,N] = A_bf16[M,K] @ W_f32[N,K]^T  (+ fused epilogue) ----------
enum { EPI_BF16 = 0, EPI_F32 = 1, EPI_RESID = 2, EPI_GELU = 3 };

template <int EPI>
__global__ __launch_bounds__(256)
void gemm_bf16_kernel(const bf16_t* __restrict__ A, const float* __restrict__ W,
                      void* __restrict__ Out, const float* __restrict__ Resid,
                      int M, int N, int K) {
  __shared__ __align__(16) bf16_t As[64 * 32];    // [m][k]
  __shared__ __align__(16) bf16_t Bs[128 * 32];   // [n][k]  (k contiguous)
  int bm = blockIdx.y, bn = blockIdx.x;
  int tid = threadIdx.x;
  int wave = tid >> 5, lane = tid & 31;
  int wm = (wave >> 2) * 32;
  int wn = (wave & 3) * 32;
  int lm = lane & 15, lg = lane >> 4;

  v8f acc[2][2] = {};
  const bf16_t* Ag = A + (long)(bm * 64) * K;
  const float*  Wg = W + (long)(bn * 128) * K;

  int ar = tid >> 2, ac = (tid & 3) * 8;          // A: row, 8 bf16 (16 B)
  int brn = tid >> 1, bkh = (tid & 1) * 16;       // B: row n, 16 f32 -> 16 bf16
  unsigned a_lds = (unsigned)(uintptr_t)(As + ar * 32 + ac);  // LDS byte offset

  for (int k0 = 0; k0 < K; k0 += 32) {
    // stage A tile (64x32 bf16) via async global->LDS DMA (ASYNCcnt path):
    // one global_load_async_to_lds_b128 per wave, 16 B per lane.
    {
      const bf16_t* gp = Ag + (long)ar * K + k0 + ac;
      asm volatile("global_load_async_to_lds_b128 %0, %1, off"
                   :: "v"(a_lds), "v"((unsigned long long)(uintptr_t)gp)
                   : "memory");
    }
    // stage + convert B tile (128x32 from fp32 weights, stored [n][k])
    {
      const float4* src = (const float4*)(Wg + (long)brn * K + k0 + bkh);
      if (k0 + 32 < K)
        __builtin_prefetch(Wg + (long)brn * K + k0 + 32 + bkh, 0, 1);
      __align__(16) bf16_t tmp[16];
#pragma unroll
      for (int i = 0; i < 4; i++) {
        float4 f = src[i];
        tmp[i * 4 + 0] = f2bf(f.x); tmp[i * 4 + 1] = f2bf(f.y);
        tmp[i * 4 + 2] = f2bf(f.z); tmp[i * 4 + 3] = f2bf(f.w);
      }
      *((uint4*)(Bs + brn * 32 + bkh))     = *((uint4*)tmp);
      *((uint4*)(Bs + brn * 32 + bkh + 8)) = *((uint4*)(tmp + 8));
    }
    asm volatile("s_wait_asynccnt 0x0" ::: "memory");
    __syncthreads();

    v16bf a0 = load_a_frag(As + (wm + lm) * 32, lg);
    v16bf a1 = load_a_frag(As + (wm + 16 + lm) * 32, lg);
    v16bf b0 = load_b_frag(Bs + (wn + lm) * 32, lg);
    v16bf b1 = load_b_frag(Bs + (wn + 16 + lm) * 32, lg);
    acc[0][0] = wmma_bf16(a0, b0, acc[0][0]);
    acc[0][1] = wmma_bf16(a0, b1, acc[0][1]);
    acc[1][0] = wmma_bf16(a1, b0, acc[1][0]);
    acc[1][1] = wmma_bf16(a1, b1, acc[1][1]);
    __syncthreads();
  }

  int rowadd = lg * 8;
#pragma unroll
  for (int mi = 0; mi < 2; mi++)
#pragma unroll
    for (int ni = 0; ni < 2; ni++)
#pragma unroll
      for (int r = 0; r < 8; r++) {
        int row = bm * 64 + wm + mi * 16 + rowadd + r;
        int col = bn * 128 + wn + ni * 16 + lm;
        long off = (long)row * N + col;
        float v = acc[mi][ni][r];
        if (EPI == EPI_BF16) {
          ((bf16_t*)Out)[off] = f2bf(v);
        } else if (EPI == EPI_F32) {
          ((float*)Out)[off] = v;
        } else if (EPI == EPI_RESID) {
          ((float*)Out)[off] = Resid[off] + v;
        } else { // EPI_GELU (exact)
          float g = 0.5f * v * (1.0f + erff(v * 0.70710678f));
          ((bf16_t*)Out)[off] = f2bf(g);
        }
      }
}

// ---------- flash attention (causal, online softmax) ----------
// grid: (T/64, H, B), 128 threads = 4 waves; wave w owns q-rows [w*16, w*16+16)
__global__ __launch_bounds__(128)
void attn_kernel(const bf16_t* __restrict__ qkv, bf16_t* __restrict__ out) {
  __shared__ __align__(16) bf16_t Qs[64 * 64];     // [q][chan]
  __shared__ __align__(16) bf16_t Ks[64 * 64];     // [key][chan]
  __shared__ __align__(16) bf16_t Vst[64 * 64];    // [chan][key] (transposed!)
  __shared__ __align__(16) bf16_t Ps[4][16 * 64];  // wave-private P tiles

  int qb = blockIdx.x, h = blockIdx.y, b = blockIdx.z;
  int tid = threadIdx.x, wave = tid >> 5, lane = tid & 31;
  int lm = lane & 15, lg = lane >> 4;
  const long rs = 3 * DMODEL;  // qkv row stride
  const bf16_t* Qg = qkv + ((long)(b * SEQ + qb * 64)) * rs + h * HD;
  const bf16_t* Kg = qkv + ((long)(b * SEQ)) * rs + DMODEL + h * HD;
  const bf16_t* Vg = qkv + ((long)(b * SEQ)) * rs + 2 * DMODEL + h * HD;

  // stage Q (fold in 1/sqrt(hd) = 0.125, exact in bf16)
  {
    int r = tid >> 1, cg = (tid & 1) * 32;
    const bf16_t* src = Qg + (long)r * rs + cg;
#pragma unroll
    for (int i = 0; i < 4; i++) {
      union { uint4 q; bf16_t e[8]; } u;
      u.q = ((const uint4*)src)[i];
#pragma unroll
      for (int e = 0; e < 8; e++) u.e[e] = f2bf(bf2f(u.e[e]) * 0.125f);
      ((uint4*)(Qs + r * 64 + cg))[i] = u.q;
    }
  }

  v8f accO[4] = {};
  float mst[8], lst[8];
#pragma unroll
  for (int r = 0; r < 8; r++) { mst[r] = -1e30f; lst[r] = 0.f; }

  int nkt = qb + 1;  // causal: key tiles 0..qb
  for (int j = 0; j < nkt; j++) {
    __syncthreads();  // previous K/V reads done (also publishes Qs on j==0)
#if HAS_TDM
    // K tile via Tensor Data Mover: 64x64 bf16 tile, row stride 3*D elems.
    if (wave == 0) {
      tdm_load_tile64x64(Kg + (long)(j * 64) * rs,
                         (unsigned)(uintptr_t)Ks, 64u, (unsigned)rs);
    }
#endif
    {
      int r = tid >> 1, cg = (tid & 1) * 32;
#if !HAS_TDM
      const bf16_t* ks = Kg + (long)(j * 64 + r) * rs + cg;
#pragma unroll
      for (int i = 0; i < 4; i++)
        ((uint4*)(Ks + r * 64 + cg))[i] = ((const uint4*)ks)[i];
#endif
      // V: stage transposed -> Vst[chan][key] so AV B-frags are contiguous
      const bf16_t* vs = Vg + (long)(j * 64 + r) * rs + cg;
#pragma unroll
      for (int i = 0; i < 4; i++) {
        union { uint4 q; bf16_t e[8]; } u;
        u.q = ((const uint4*)vs)[i];
#pragma unroll
        for (int e = 0; e < 8; e++) Vst[(cg + i * 8 + e) * 64 + r] = u.e[e];
      }
    }
#if HAS_TDM
    if (wave == 0) __builtin_amdgcn_s_wait_tensorcnt(0);
#endif
    __syncthreads();

    // S = Qw(16x64) @ K^T: B(k=chan, n=key) = Ks[key][chan] (chan contiguous)
    v8f s[4] = {};
#pragma unroll
    for (int kc = 0; kc < 2; kc++) {
      v16bf af = load_a_frag(Qs + (wave * 16 + lm) * 64 + kc * 32, lg);
#pragma unroll
      for (int ni = 0; ni < 4; ni++) {
        v16bf bf = load_b_frag(Ks + (ni * 16 + lm) * 64 + kc * 32, lg);
        s[ni] = wmma_bf16(af, bf, s[ni]);
      }
    }
    // causal mask (diagonal tile only)
    if (j == qb) {
#pragma unroll
      for (int ni = 0; ni < 4; ni++)
#pragma unroll
        for (int r = 0; r < 8; r++) {
          int qrow = wave * 16 + lg * 8 + r;
          if (ni * 16 + lm > qrow) s[ni][r] = -__builtin_inff();
        }
    }
    // online softmax stats (row = 16-lane half x VGPR index r)
#pragma unroll
    for (int r = 0; r < 8; r++) {
      float v = fmaxf(fmaxf(s[0][r], s[1][r]), fmaxf(s[2][r], s[3][r]));
#pragma unroll
      for (int off = 8; off >= 1; off >>= 1) v = fmaxf(v, __shfl_xor(v, off, 32));
      float mnew = fmaxf(mst[r], v);
      float corr = __expf(mst[r] - mnew);
      mst[r] = mnew;
      lst[r] *= corr;
#pragma unroll
      for (int ni = 0; ni < 4; ni++) accO[ni][r] *= corr;
    }
    // P = exp(S-m); spill to wave-private LDS for A-fragment reshape
#pragma unroll
    for (int ni = 0; ni < 4; ni++)
#pragma unroll
      for (int r = 0; r < 8; r++) {
        float sv = s[ni][r];
        float p = (sv == -__builtin_inff()) ? 0.f : __expf(sv - mst[r]);
        s[ni][r] = p;
        Ps[wave][(lg * 8 + r) * 64 + ni * 16 + lm] = f2bf(p);
      }
#pragma unroll
    for (int r = 0; r < 8; r++) {
      float v = s[0][r] + s[1][r] + s[2][r] + s[3][r];
#pragma unroll
      for (int off = 8; off >= 1; off >>= 1) v += __shfl_xor(v, off, 32);
      lst[r] += v;
    }
    // O += P(16x64) @ V(64x64): B(k=key, n=chan) = Vst[chan][key], contiguous
#pragma unroll
    for (int kc = 0; kc < 2; kc++) {
      v16bf af = load_a_frag(Ps[wave] + lm * 64 + kc * 32, lg);
#pragma unroll
      for (int ni = 0; ni < 4; ni++) {
        v16bf bf = load_b_frag(Vst + (ni * 16 + lm) * 64 + kc * 32, lg);
        accO[ni] = wmma_bf16(af, bf, accO[ni]);
      }
    }
  }
  // finalize O/l -> attn_out bf16 [2048][1024]
  bf16_t* og = out + ((long)(b * SEQ + qb * 64)) * DMODEL + h * HD;
#pragma unroll
  for (int ni = 0; ni < 4; ni++)
#pragma unroll
    for (int r = 0; r < 8; r++) {
      int row = wave * 16 + lg * 8 + r;
      int c = ni * 16 + lm;
      og[(long)row * DMODEL + c] = f2bf(accO[ni][r] / lst[r]);
    }
}

// ---------- host orchestration ----------
template <int EPI>
static void launch_gemm(const bf16_t* A, const float* W, void* out,
                        const float* resid, int M, int N, int K, hipStream_t s) {
  dim3 grid(N / 128, M / 64);
  gemm_bf16_kernel<EPI><<<grid, 256, 0, s>>>(A, W, out, resid, M, N, K);
}

extern "C" void kernel_launch(void* const* d_in, const int* in_sizes, int n_in,
                              void* d_out, int out_size, void* d_ws, size_t ws_size,
                              hipStream_t stream) {
  const int*   idx  = (const int*)d_in[0];
  const float* tok  = (const float*)d_in[1];
  const float* pos  = (const float*)d_in[2];
  const float* ln1w = (const float*)d_in[3];
  const float* ln1b = (const float*)d_in[4];
  const float* qkvw = (const float*)d_in[5];
  const float* outw = (const float*)d_in[6];
  const float* ln2w = (const float*)d_in[7];
  const float* ln2b = (const float*)d_in[8];
  const float* f1w  = (const float*)d_in[9];
  const float* f2w  = (const float*)d_in[10];
  const float* lnfw = (const float*)d_in[11];
  const float* lnfb = (const float*)d_in[12];

  char* p = (char*)d_ws;
  float*  x      = (float*)p;   p += (size_t)NTOK * DMODEL * 4;
  bf16_t* hbf    = (bf16_t*)p;  p += (size_t)NTOK * DMODEL * 2;
  bf16_t* qkvbf  = (bf16_t*)p;  p += (size_t)NTOK * 3 * DMODEL * 2;
  bf16_t* attnbf = (bf16_t*)p;  p += (size_t)NTOK * DMODEL * 2;
  bf16_t* ffnbf  = (bf16_t*)p;  p += (size_t)NTOK * DFFN * 2;

  embed_kernel<<<NTOK, 256, 0, stream>>>(idx, tok, pos, x);

  for (int l = 0; l < NLAYER; l++) {
    ln_bf16_kernel<<<NTOK, 256, 0, stream>>>(x, ln1w + l * DMODEL, ln1b + l * DMODEL, hbf);
    launch_gemm<EPI_BF16>(hbf, qkvw + (size_t)l * 3 * DMODEL * DMODEL, qkvbf,
                          nullptr, NTOK, 3 * DMODEL, DMODEL, stream);
    attn_kernel<<<dim3(SEQ / 64, NHEAD, 2), 128, 0, stream>>>(qkvbf, attnbf);
    launch_gemm<EPI_RESID>(attnbf, outw + (size_t)l * DMODEL * DMODEL, x,
                           x, NTOK, DMODEL, DMODEL, stream);
    ln_bf16_kernel<<<NTOK, 256, 0, stream>>>(x, ln2w + l * DMODEL, ln2b + l * DMODEL, hbf);
    launch_gemm<EPI_GELU>(hbf, f1w + (size_t)l * DFFN * DMODEL, ffnbf,
                          nullptr, NTOK, DFFN, DMODEL, stream);
    launch_gemm<EPI_RESID>(ffnbf, f2w + (size_t)l * DMODEL * DFFN, x,
                           x, NTOK, DMODEL, DFFN, stream);
  }

  ln_bf16_kernel<<<NTOK, 256, 0, stream>>>(x, lnfw, lnfb, hbf);
  launch_gemm<EPI_F32>(hbf, tok, d_out, nullptr, NTOK, VOCAB, DMODEL, stream);
}
